// ParallelHybridReasoningLayer_90752658964695
// MI455X (gfx1250) — compile-verified, matrix operand order checked by reading, not run on hardware
//
#include <hip/hip_runtime.h>
#include <hip/hip_bf16.h>
#include <math.h>

// ---------------- problem constants ----------------
constexpr int NB = 4;      // batch
constexpr int TT = 2048;   // sequence
constexpr int Dm = 256;    // model dim
constexpr int NH = 4;      // heads
constexpr int NG = 4;      // global tokens
constexpr int KW = 8;      // conv taps
constexpr long NTD = (long)NB * TT * Dm;

typedef __bf16 bf16;
typedef __attribute__((ext_vector_type(16))) __bf16 v16bf;
typedef __attribute__((ext_vector_type(8)))  __bf16 v8bf;
typedef __attribute__((ext_vector_type(8)))  float  v8f;
typedef __attribute__((ext_vector_type(4)))  unsigned int v4u;
typedef __attribute__((ext_vector_type(8)))  int v8i;
typedef __attribute__((ext_vector_type(4)))  int v4i;

union FragAB { v16bf v; v8bf h[2]; };

__device__ __forceinline__ float gelu_exact(float x) {
    return 0.5f * x * (1.0f + erff(x * 0.70710678118654752f));
}

__device__ __forceinline__ v8f wmma_bf16(const FragAB& a, const FragAB& b, v8f c) {
    return __builtin_amdgcn_wmma_f32_16x16x32_bf16(
               false, a.v, false, b.v, (short)0, c, false, false);
}

__device__ __forceinline__ float epilogue(float v, int act) {
    if (act == 1)      v = gelu_exact(v);
    else if (act == 2) v = 1.0f / (1.0f + __expf(-v));
    return v;
}

// ======================================================================
// WMMA GEMM (small-M fallback): one wave per 16x16 tile, operands from
// global. C[M,N] = act(A[M,Kd] @ W[N,Kd]^T + bias).
// ======================================================================
__global__ void k_gemm(const bf16* __restrict__ A, int lda,
                       const bf16* __restrict__ W, int ldw,
                       const float* __restrict__ bias,
                       float* __restrict__ Cf, bf16* __restrict__ Ch, int ldc,
                       int Kd, int act)
{
    const int lane = threadIdx.x;
    const int n0 = blockIdx.x * 16;
    const int m0 = blockIdx.y * 16;
    const int l = lane & 15;
    const int half = lane >> 4;

    const bf16* ap = A + (size_t)(m0 + l) * lda + half * 8;
    const bf16* wp = W + (size_t)(n0 + l) * ldw + half * 8;

    v8f acc = {};
    for (int k0 = 0; k0 < Kd; k0 += 32) {
        __builtin_prefetch(ap + k0 + 64, 0, 0);   // global_prefetch_b8
        FragAB a, b;
        a.h[0] = *(const v8bf*)(ap + k0);
        a.h[1] = *(const v8bf*)(ap + k0 + 16);
        b.h[0] = *(const v8bf*)(wp + k0);
        b.h[1] = *(const v8bf*)(wp + k0 + 16);
        acc = wmma_bf16(a, b, acc);
    }
    const float bv = bias ? bias[n0 + l] : 0.0f;
#pragma unroll
    for (int r = 0; r < 8; ++r) {
        const int row = m0 + r + half * 8;
        const float v = epilogue(acc[r] + bv, act);
        const size_t o = (size_t)row * ldc + n0 + l;
        if (Cf) Cf[o] = v;
        if (Ch) Ch[o] = (bf16)v;
    }
}

// ======================================================================
// WMMA GEMM with Tensor-Data-Mover staging of the weight panel.
// Block = 128 threads (4 waves). Block tile: 64 rows (M) x 16 cols (N).
// Per 256-wide K slab, wave 0 issues one TDM descriptor that DMAs the
// 16 x 256 bf16 W panel into LDS (tensor_load_to_lds, TENSORcnt), then
// all 4 waves consume it via ds_load_b128 while streaming their private
// A fragments from global. Requires M % 64 == 0, Kd % 256 == 0.
// ======================================================================
constexpr int KS = 256;   // K slab (elements)

__global__ void __launch_bounds__(128)
k_gemm_tdm(const bf16* __restrict__ A, int lda,
           const bf16* __restrict__ W, int ldw,
           const float* __restrict__ bias,
           float* __restrict__ Cf, bf16* __restrict__ Ch, int ldc,
           int Kd, int act)
{
    __shared__ __align__(16) bf16 sw[16 * KS];    // 8 KB weight panel slab
    const int lane = threadIdx.x & 31;
    const int wv   = threadIdx.x >> 5;            // wave id 0..3
    const int n0 = blockIdx.x * 16;
    const int m0 = blockIdx.y * 64 + wv * 16;
    const int l = lane & 15;
    const int half = lane >> 4;

    const bf16* ap  = A + (size_t)(m0 + l) * lda + half * 8;
    const bf16* swp = sw + l * KS + half * 8;

    v8f acc = {};
    for (int ks0 = 0; ks0 < Kd; ks0 += KS) {
        if (threadIdx.x < 32) {
            // ---- TDM descriptor: 2D tile, 16 rows x KS cols, 2-byte elems
            const unsigned long long ga =
                (unsigned long long)(uintptr_t)(W + (size_t)n0 * ldw + ks0);
            const unsigned int ldsa = (unsigned int)(uintptr_t)(void*)sw;
            // group0: count=1 | lds_addr | global_addr[56:0] | type=2
            v4u g0 = { 1u,
                       ldsa,
                       (unsigned int)ga,
                       ((unsigned int)(ga >> 32) & 0x01FFFFFFu) | 0x80000000u };
            // group1: data_size=1(2B); tensor_dim0=KS; tensor_dim1=16;
            //         tile_dim0=KS; tile_dim1=16; tensor_dim0_stride=ldw
            v8i g1 = { (int)0x00010000,
                       (int)(KS << 16),        // dim0[15:0]  in [31:16]
                       (int)(16 << 16),        // dim1[15:0]  in [31:16]
                       (int)(KS << 16),        // tile_dim0   in [31:16]
                       (int)16,                // tile_dim1
                       (int)ldw,               // dim0_stride[31:0]
                       0, 0 };
            v4i gz = { 0, 0, 0, 0 };
#if defined(__clang_major__) && (__clang_major__ >= 23)
            v8i gz8 = { 0, 0, 0, 0, 0, 0, 0, 0 };
            __builtin_amdgcn_tensor_load_to_lds(g0, g1, gz, gz, gz8, 0);
#else
            __builtin_amdgcn_tensor_load_to_lds(g0, g1, gz, gz, 0);
#endif
            __builtin_amdgcn_s_wait_tensorcnt(0);
        }
        __syncthreads();
#pragma unroll
        for (int k0 = 0; k0 < KS; k0 += 32) {
            __builtin_prefetch(ap + ks0 + k0 + 64, 0, 0);
            FragAB a, b;
            a.h[0] = *(const v8bf*)(ap + ks0 + k0);
            a.h[1] = *(const v8bf*)(ap + ks0 + k0 + 16);
            b.h[0] = *(const v8bf*)(swp + k0);        // ds_load_b128
            b.h[1] = *(const v8bf*)(swp + k0 + 16);
            acc = wmma_bf16(a, b, acc);
        }
        __syncthreads();
    }
    const float bv = bias ? bias[n0 + l] : 0.0f;
#pragma unroll
    for (int r = 0; r < 8; ++r) {
        const int row = m0 + r + half * 8;
        const float v = epilogue(acc[r] + bv, act);
        const size_t o = (size_t)row * ldc + n0 + l;
        if (Cf) Cf[o] = v;
        if (Ch) Ch[o] = (bf16)v;
    }
}

// ======================================================================
// Windowed (W=64) multi-head attention over qkv (bf16, row stride 3D).
// One wave per (batch, head, 16-query tile). Online softmax + WMMA.
// ======================================================================
__global__ void k_med_attn(const bf16* __restrict__ qkv, bf16* __restrict__ ctx)
{
    const int lane = threadIdx.x;
    const int qt = blockIdx.x;
    const int h  = blockIdx.y;
    const int b  = blockIdx.z;
    const int q0 = qt * 16;
    const int l = lane & 15;
    const int half = lane >> 4;
    const size_t rs = 3 * Dm;

    const bf16* Q  = qkv + (size_t)b * TT * rs + (size_t)h * 64;
    const bf16* Kp = Q + Dm;
    const bf16* Vp = Q + 2 * Dm;

    FragAB qf[2];
    {
        const bf16* qp = Q + (size_t)(q0 + l) * rs + half * 8;
        qf[0].h[0] = *(const v8bf*)(qp);
        qf[0].h[1] = *(const v8bf*)(qp + 16);
        qf[1].h[0] = *(const v8bf*)(qp + 32);
        qf[1].h[1] = *(const v8bf*)(qp + 48);
    }

    float mi[8], li[8];
    v8f o[4] = {};
#pragma unroll
    for (int r = 0; r < 8; ++r) { mi[r] = -INFINITY; li[r] = 0.0f; }

    int kstart = q0 - 63; if (kstart < 0) kstart = 0;
    for (int ks = (kstart / 16) * 16; ks <= q0; ks += 16) {
        FragAB kf[2];
        const bf16* kp = Kp + (size_t)(ks + l) * rs + half * 8;
        kf[0].h[0] = *(const v8bf*)(kp);
        kf[0].h[1] = *(const v8bf*)(kp + 16);
        kf[1].h[0] = *(const v8bf*)(kp + 32);
        kf[1].h[1] = *(const v8bf*)(kp + 48);

        v8f s = {};
        s = wmma_bf16(qf[0], kf[0], s);
        s = wmma_bf16(qf[1], kf[1], s);

        float p[8];
#pragma unroll
        for (int r = 0; r < 8; ++r) {
            const int qrow = q0 + r + half * 8;
            const int kcol = ks + l;
            float sv = s[r] * 0.125f;        // 1/sqrt(64)
            const int diff = qrow - kcol;
            if (diff < 0 || diff >= 64) sv = -INFINITY;
            float rm = sv;                    // row max over 16-lane half
            rm = fmaxf(rm, __shfl_xor(rm, 1, 32));
            rm = fmaxf(rm, __shfl_xor(rm, 2, 32));
            rm = fmaxf(rm, __shfl_xor(rm, 4, 32));
            rm = fmaxf(rm, __shfl_xor(rm, 8, 32));
            const float nm = fmaxf(mi[r], rm);
            float pv, alpha;
            if (nm == -INFINITY) { pv = 0.0f; alpha = 1.0f; }
            else {
                pv = (sv == -INFINITY) ? 0.0f : __expf(sv - nm);
                alpha = (mi[r] == -INFINITY) ? 0.0f : __expf(mi[r] - nm);
            }
            float rsum = pv;
            rsum += __shfl_xor(rsum, 1, 32);
            rsum += __shfl_xor(rsum, 2, 32);
            rsum += __shfl_xor(rsum, 4, 32);
            rsum += __shfl_xor(rsum, 8, 32);
            li[r] = li[r] * alpha + rsum;
            mi[r] = nm;
            p[r] = pv;
            o[0][r] *= alpha; o[1][r] *= alpha; o[2][r] *= alpha; o[3][r] *= alpha;
        }

        // Transpose P from C-layout into an A-fragment (K=0..15 valid, rest 0)
        float pv8[8];
#pragma unroll
        for (int r = 0; r < 8; ++r) {
#pragma unroll
            for (int j = 0; j < 8; ++j) {
                const int src = half * 8 + j + ((l & 8) ? 16 : 0);
                const float t = __shfl(p[r], src, 32);
                if ((l & 7) == r) pv8[j] = t;
            }
        }
        FragAB pa;
#pragma unroll
        for (int j = 0; j < 8; ++j) { pa.v[j] = (bf16)pv8[j]; pa.v[j + 8] = (bf16)0.0f; }

        // O += P @ V  (4 hd tiles of 16 columns)
#pragma unroll
        for (int ht = 0; ht < 4; ++ht) {
            FragAB vf;
#pragma unroll
            for (int j = 0; j < 8; ++j) {
                const int key = ks + half * 8 + j;
                vf.v[j] = Vp[(size_t)key * rs + ht * 16 + l];
                vf.v[j + 8] = (bf16)0.0f;
            }
            o[ht] = wmma_bf16(pa, vf, o[ht]);
        }
    }

#pragma unroll
    for (int r = 0; r < 8; ++r) {
        const int row = q0 + r + half * 8;
        const float inv = (li[r] > 0.0f) ? 1.0f / li[r] : 0.0f;
#pragma unroll
        for (int ht = 0; ht < 4; ++ht)
            ctx[(size_t)(b * TT + row) * Dm + h * 64 + ht * 16 + l] = (bf16)(o[ht][r] * inv);
    }
}

// ======================================================================
// Global attention: block per (b,g,h), full-length softmax over T keys.
// ======================================================================
__global__ void k_gattn(const float* __restrict__ qg, const bf16* __restrict__ kvg,
                        bf16* __restrict__ gc)
{
    const int blk = blockIdx.x;
    const int h = blk & 3, g = (blk >> 2) & 3, b = blk >> 4;
    const int tid = threadIdx.x;              // 64 threads
    __shared__ float qs[64];
    __shared__ float sc[TT];
    __shared__ float red[64];

    qs[tid] = qg[(size_t)(b * NG + g) * Dm + h * 64 + tid];
    __syncthreads();

    const bf16* kb = kvg + (size_t)b * TT * (2 * Dm) + h * 64;
    const bf16* vb = kb + Dm;

    float lm = -INFINITY;
    for (int key = tid; key < TT; key += 64) {
        float dot = 0.0f;
        const bf16* kr = kb + (size_t)key * (2 * Dm);
        for (int j = 0; j < 64; ++j) dot += qs[j] * (float)kr[j];
        dot *= 0.125f;
        sc[key] = dot;
        lm = fmaxf(lm, dot);
    }
    red[tid] = lm; __syncthreads();
    for (int s = 32; s > 0; s >>= 1) { if (tid < s) red[tid] = fmaxf(red[tid], red[tid + s]); __syncthreads(); }
    const float m = red[0]; __syncthreads();

    float ls = 0.0f;
    for (int key = tid; key < TT; key += 64) ls += __expf(sc[key] - m);
    red[tid] = ls; __syncthreads();
    for (int s = 32; s > 0; s >>= 1) { if (tid < s) red[tid] += red[tid + s]; __syncthreads(); }
    const float denom = red[0];

    float acc = 0.0f;                          // thread owns hd column tid
    for (int key = 0; key < TT; ++key)
        acc += __expf(sc[key] - m) * (float)vb[(size_t)key * (2 * Dm) + tid];
    gc[(size_t)(b * NG + g) * Dm + h * 64 + tid] = (bf16)(acc / denom);
}

// ---------------- elementwise / helper kernels ----------------
__global__ void k_cast(const float* __restrict__ in, bf16* __restrict__ out, long n) {
    long i = (long)blockIdx.x * blockDim.x + threadIdx.x;
    const long st = (long)gridDim.x * blockDim.x;
    for (; i < n; i += st) out[i] = (bf16)in[i];
}

__global__ void k_conv(const float* __restrict__ x, const float* __restrict__ w,
                       const float* __restrict__ bi, float* __restrict__ lh) {
    long i = (long)blockIdx.x * blockDim.x + threadIdx.x;
    const long st = (long)gridDim.x * blockDim.x;
    for (; i < NTD; i += st) {
        const int d = (int)(i % Dm);
        const long bt = i / Dm;
        const int t = (int)(bt % TT);
        const int b = (int)(bt / TT);
        float s = bi[d];
#pragma unroll
        for (int j = 0; j < KW; ++j) {
            const int tt = t - (KW - 1) + j;
            if (tt >= 0) s += x[((size_t)b * TT + tt) * Dm + d] * w[d * KW + j];
        }
        lh[i] = s;
    }
}

__global__ void k_cat2(const bf16* __restrict__ xb, const float* __restrict__ med,
                       bf16* __restrict__ cat2) {
    long i = (long)blockIdx.x * blockDim.x + threadIdx.x;
    const long st = (long)gridDim.x * blockDim.x;
    for (; i < NTD; i += st) {
        const long row = i / Dm; const int d = (int)(i % Dm);
        cat2[row * (2 * Dm) + d] = xb[i];
        cat2[row * (2 * Dm) + Dm + d] = (bf16)med[i];
    }
}

__global__ void k_cat3(const float* __restrict__ lh, const float* __restrict__ med,
                       const float* __restrict__ fg, const float* __restrict__ gmean,
                       bf16* __restrict__ cat3) {
    long i = (long)blockIdx.x * blockDim.x + threadIdx.x;
    const long st = (long)gridDim.x * blockDim.x;
    for (; i < NTD; i += st) {
        const long row = i / Dm; const int d = (int)(i % Dm);
        const int b = (int)(row / TT);
        cat3[row * (3 * Dm) + d] = (bf16)lh[i];
        cat3[row * (3 * Dm) + Dm + d] = (bf16)(med[i] * fg[i]);
        cat3[row * (3 * Dm) + 2 * Dm + d] = (bf16)gmean[(size_t)b * Dm + d];
    }
}

__global__ void k_cs(const float* __restrict__ x1, const float* __restrict__ pg,
                     const float* __restrict__ ema, bf16* __restrict__ cs) {
    long i = (long)blockIdx.x * blockDim.x + threadIdx.x;
    const long st = (long)gridDim.x * blockDim.x;
    for (; i < NTD; i += st) {
        const long row = i / Dm; const int d = (int)(i % Dm);
        cs[row * (2 * Dm) + d] = (bf16)(x1[i] + 0.3f * pg[i]);
        cs[row * (2 * Dm) + Dm + d] = (bf16)ema[i];
    }
}

__global__ void k_fi(const float* __restrict__ x1, const float* __restrict__ sg,
                     bf16* __restrict__ fi) {
    long i = (long)blockIdx.x * blockDim.x + threadIdx.x;
    const long st = (long)gridDim.x * blockDim.x;
    for (; i < NTD; i += st) fi[i] = (bf16)(x1[i] + 0.3f * sg[i]);
}

__global__ void k_gq(const float* __restrict__ gt, bf16* __restrict__ gq) {
    const int i = blockIdx.x * blockDim.x + threadIdx.x;
    if (i < NB * NG * Dm) {
        const int g = (i / Dm) % NG; const int d = i % Dm;
        gq[i] = (bf16)gt[g * Dm + d];
    }
}

__global__ void k_gmean(const float* __restrict__ gcp, float* __restrict__ gm) {
    const int i = blockIdx.x * blockDim.x + threadIdx.x;
    if (i < NB * Dm) {
        const int b = i / Dm, d = i % Dm;
        float s = 0.0f;
        for (int g = 0; g < NG; ++g) s += gcp[(size_t)(b * NG + g) * Dm + d];
        gm[i] = s * (1.0f / NG);
    }
}

__global__ void k_ema(const float* __restrict__ x1, float* __restrict__ ema) {
    const int i = blockIdx.x * blockDim.x + threadIdx.x;
    if (i >= NB * Dm) return;
    const int b = i / Dm, d = i % Dm;
    const float* xp = x1 + (size_t)b * TT * Dm + d;
    float* ep = ema + (size_t)b * TT * Dm + d;
    float c = 0.0f;
    for (int t = 0; t < TT; ++t) {
        c = 0.9f * c + 0.1f * xp[(size_t)t * Dm];
        ep[(size_t)t * Dm] = c;
    }
}

__global__ void k_cumsum(const float* __restrict__ ss, const float* __restrict__ deltas,
                         float* __restrict__ states) {
    const int i = blockIdx.x * blockDim.x + threadIdx.x;
    if (i >= NB * Dm) return;
    const int b = i / Dm, d = i % Dm;
    const float* dp = deltas + (size_t)b * TT * Dm + d;
    float* sp = states + (size_t)b * TT * Dm + d;
    float c = ss[i];
    for (int t = 0; t < TT; ++t) {
        c += 0.5f * dp[(size_t)t * Dm];
        sp[(size_t)t * Dm] = c;
    }
}

__global__ void k_ln(const float* __restrict__ in, long istride,
                     const float* __restrict__ res,
                     const float* __restrict__ g, const float* __restrict__ be,
                     float* __restrict__ outf, bf16* __restrict__ outh, long ostride) {
    const int row = blockIdx.x;
    const int d = threadIdx.x;                 // blockDim == 256
    __shared__ float sm[256];
    float v = in[(size_t)row * istride + d];
    if (res) v += res[(size_t)row * Dm + d];
    sm[d] = v; __syncthreads();
    for (int s = 128; s > 0; s >>= 1) { if (d < s) sm[d] += sm[d + s]; __syncthreads(); }
    const float mean = sm[0] * (1.0f / Dm); __syncthreads();
    const float dv = v - mean;
    sm[d] = dv * dv; __syncthreads();
    for (int s = 128; s > 0; s >>= 1) { if (d < s) sm[d] += sm[d + s]; __syncthreads(); }
    const float var = sm[0] * (1.0f / Dm);
    const float y = dv * rsqrtf(var + 1e-5f) * g[d] + be[d];
    if (outf) outf[(size_t)row * ostride + d] = y;
    if (outh) outh[(size_t)row * ostride + d] = (bf16)y;
}

__global__ void k_traj(const float* __restrict__ ema, float* __restrict__ out) {
    const int i = blockIdx.x * blockDim.x + threadIdx.x;
    if (i < NB * Dm) {
        const int b = i / Dm, d = i % Dm;
        out[i] = ema[((size_t)b * TT + TT - 1) * Dm + d];
    }
}

// ======================================================================
extern "C" void kernel_launch(void* const* d_in, const int* in_sizes, int n_in,
                              void* d_out, int out_size, void* d_ws, size_t ws_size,
                              hipStream_t stream) {
    const float* x        = (const float*)d_in[0];
    const float* seqstate = (const float*)d_in[1];
    const float* conv_w   = (const float*)d_in[3];
    const float* conv_b   = (const float*)d_in[4];
    const float* med_in_w = (const float*)d_in[5];
    const float* med_in_b = (const float*)d_in[6];
    const float* med_out_w= (const float*)d_in[7];
    const float* med_out_b= (const float*)d_in[8];
    const float* forget_w = (const float*)d_in[9];
    const float* forget_b = (const float*)d_in[10];
    const float* gtok     = (const float*)d_in[11];
    const float* glob_in_w= (const float*)d_in[12];
    const float* glob_in_b= (const float*)d_in[13];
    const float* glob_out_w=(const float*)d_in[14];
    const float* glob_out_b=(const float*)d_in[15];
    const float* combine_w= (const float*)d_in[16];
    const float* combine_b= (const float*)d_in[17];
    const float* p2s_w    = (const float*)d_in[18];
    const float* p2s_b    = (const float*)d_in[19];
    const float* s2p_w    = (const float*)d_in[20];
    const float* s2p_b    = (const float*)d_in[21];
    const float* t1_w     = (const float*)d_in[22];
    const float* t1_b     = (const float*)d_in[23];
    const float* t2_w     = (const float*)d_in[24];
    const float* t2_b     = (const float*)d_in[25];
    const float* ffn1_w   = (const float*)d_in[26];
    const float* ffn1_b   = (const float*)d_in[27];
    const float* ffn2_w   = (const float*)d_in[28];
    const float* ffn2_b   = (const float*)d_in[29];
    const float* ln1_g = (const float*)d_in[30]; const float* ln1_b = (const float*)d_in[31];
    const float* ln2_g = (const float*)d_in[32]; const float* ln2_b = (const float*)d_in[33];
    const float* ln3_g = (const float*)d_in[34]; const float* ln3_b = (const float*)d_in[35];
    const float* ln4_g = (const float*)d_in[36]; const float* ln4_b = (const float*)d_in[37];

    float* dout = (float*)d_out;

    // ---------------- workspace layout ----------------
    char* base = (char*)d_ws;
    size_t off = 0;
    auto alloc = [&](size_t bytes) -> void* {
        void* p = base + off;
        off += (bytes + 255) & ~(size_t)255;
        return p;
    };
    // f32 buffers (with lifetime-based aliasing)
    float* lh       = (float*)alloc(NTD * 4);   // -> reused as states
    float* med      = (float*)alloc(NTD * 4);   // -> reused as pg
    float* fg       = (float*)alloc(NTD * 4);   // -> reused as deltas, then ffn
    float* comb     = (float*)alloc(NTD * 4);   // -> reused as sg
    float* x1       = (float*)alloc(NTD * 4);
    float* ema      = (float*)alloc(NTD * 4);
    float* states_ln= (float*)alloc(NTD * 4);
    float* qg16     = (float*)alloc((size_t)NB * NG * Dm * 4);
    float* gcp      = (float*)alloc((size_t)NB * NG * Dm * 4);
    float* gmean    = (float*)alloc((size_t)NB * Dm * 4);
    float* states = lh;
    float* pg     = med;
    float* deltas = fg;
    float* sg     = comb;
    float* ffn    = fg;

    // bf16 buffers (aliased by lifetime)
    bf16* x_bf   = (bf16*)alloc(NTD * 2);       // -> reused as states_bf
    bf16* qkv_bf = (bf16*)alloc(3 * NTD * 2);   // -> reused as cat3
    bf16* ctx_bf = (bf16*)alloc(NTD * 2);       // -> reused as x1_bf
    bf16* cat2   = (bf16*)alloc(2 * NTD * 2);   // -> reused as cs
    bf16* kvg    = (bf16*)alloc(2 * NTD * 2);   // -> reused as h_bf
    bf16* fi_bf  = (bf16*)alloc(NTD * 2);
    bf16* hf_bf  = (bf16*)alloc(4 * NTD * 2);
    bf16* gq_bf  = (bf16*)alloc((size_t)NB * NG * Dm * 2);
    bf16* gc_bf  = (bf16*)alloc((size_t)NB * NG * Dm * 2);
    bf16* cat3      = qkv_bf;
    bf16* x1_bf     = ctx_bf;
    bf16* cs        = cat2;
    bf16* h_bf      = kvg;
    bf16* states_bf = x_bf;

    // bf16 weights
    bf16* med_in_wh  = (bf16*)alloc((size_t)3 * Dm * Dm * 2);
    bf16* med_out_wh = (bf16*)alloc((size_t)Dm * Dm * 2);
    bf16* forget_wh  = (bf16*)alloc((size_t)Dm * 2 * Dm * 2);
    bf16* glob_in_wh = (bf16*)alloc((size_t)3 * Dm * Dm * 2);
    bf16* glob_out_wh= (bf16*)alloc((size_t)Dm * Dm * 2);
    bf16* combine_wh = (bf16*)alloc((size_t)Dm * 3 * Dm * 2);
    bf16* p2s_wh     = (bf16*)alloc((size_t)Dm * Dm * 2);
    bf16* s2p_wh     = (bf16*)alloc((size_t)Dm * Dm * 2);
    bf16* t1_wh      = (bf16*)alloc((size_t)2 * Dm * 2 * Dm * 2);
    bf16* t2_wh      = (bf16*)alloc((size_t)Dm * 2 * Dm * 2);
    bf16* ffn1_wh    = (bf16*)alloc((size_t)4 * Dm * Dm * 2);
    bf16* ffn2_wh    = (bf16*)alloc((size_t)Dm * 4 * Dm * 2);

    const int MT = NB * TT;            // 8192 rows
    auto cast = [&](const float* in, bf16* outp, long n) {
        int blocks = (int)((n + 255) / 256); if (blocks > 8192) blocks = 8192;
        k_cast<<<dim3(blocks), dim3(256), 0, stream>>>(in, outp, n);
    };
    auto gemm = [&](const bf16* A, int lda, const bf16* W, int ldw, const float* bias,
                    float* Cf, bf16* Ch, int ldc, int M, int N, int Kd, int act) {
        if ((M % 64) == 0 && (Kd % KS) == 0) {
            // TDM-staged weight panel, 4 waves per block share it from LDS
            k_gemm_tdm<<<dim3(N / 16, M / 64), dim3(128), 0, stream>>>(
                A, lda, W, ldw, bias, Cf, Ch, ldc, Kd, act);
        } else {
            k_gemm<<<dim3(N / 16, M / 16), dim3(32), 0, stream>>>(
                A, lda, W, ldw, bias, Cf, Ch, ldc, Kd, act);
        }
    };
    const dim3 EW((unsigned)((NTD + 255) / 256)), ET(256);

    // ---------------- pipeline ----------------
    cast(x, x_bf, NTD);
    cast(med_in_w,  med_in_wh,  (long)3 * Dm * Dm);
    cast(med_out_w, med_out_wh, (long)Dm * Dm);
    cast(forget_w,  forget_wh,  (long)Dm * 2 * Dm);
    cast(glob_in_w, glob_in_wh, (long)3 * Dm * Dm);
    cast(glob_out_w,glob_out_wh,(long)Dm * Dm);
    cast(combine_w, combine_wh, (long)Dm * 3 * Dm);
    cast(p2s_w,     p2s_wh,     (long)Dm * Dm);
    cast(s2p_w,     s2p_wh,     (long)Dm * Dm);
    cast(t1_w,      t1_wh,      (long)2 * Dm * 2 * Dm);
    cast(t2_w,      t2_wh,      (long)Dm * 2 * Dm);
    cast(ffn1_w,    ffn1_wh,    (long)4 * Dm * Dm);
    cast(ffn2_w,    ffn2_wh,    (long)Dm * 4 * Dm);

    // local depthwise causal conv
    k_conv<<<dim3(4096), ET, 0, stream>>>(x, conv_w, conv_b, lh);

    // medium (windowed) attention branch
    gemm(x_bf, Dm, med_in_wh, Dm, med_in_b, nullptr, qkv_bf, 3 * Dm, MT, 3 * Dm, Dm, 0);
    k_med_attn<<<dim3(TT / 16, NH, NB), dim3(32), 0, stream>>>(qkv_bf, ctx_bf);
    gemm(ctx_bf, Dm, med_out_wh, Dm, med_out_b, med, nullptr, Dm, MT, Dm, Dm, 0);
    k_cat2<<<EW, ET, 0, stream>>>(x_bf, med, cat2);
    gemm(cat2, 2 * Dm, forget_wh, 2 * Dm, forget_b, fg, nullptr, Dm, MT, Dm, 2 * Dm, 2);

    // global attention branch
    k_gq<<<dim3((NB * NG * Dm + 255) / 256), ET, 0, stream>>>(gtok, gq_bf);
    gemm(gq_bf, Dm, glob_in_wh, Dm, glob_in_b, qg16, nullptr, Dm, NB * NG, Dm, Dm, 0);
    gemm(x_bf, Dm, glob_in_wh + (size_t)Dm * Dm, Dm, glob_in_b + Dm,
         nullptr, kvg, 2 * Dm, MT, 2 * Dm, Dm, 0);
    k_gattn<<<dim3(NB * NG * NH), dim3(64), 0, stream>>>(qg16, kvg, gc_bf);
    gemm(gc_bf, Dm, glob_out_wh, Dm, glob_out_b, gcp, nullptr, Dm, NB * NG, Dm, Dm, 0);
    k_gmean<<<dim3((NB * Dm + 255) / 256), ET, 0, stream>>>(gcp, gmean);

    // combine + LN1
    k_cat3<<<EW, ET, 0, stream>>>(lh, med, fg, gmean, cat3);
    gemm(cat3, 3 * Dm, combine_wh, 3 * Dm, combine_b, comb, nullptr, Dm, MT, Dm, 3 * Dm, 0);
    k_ln<<<dim3(MT), ET, 0, stream>>>(comb, Dm, x, ln1_g, ln1_b, x1, x1_bf, Dm);

    // sequential branch: EMA, p2s, t1/t2, cumsum, LN3, s2p
    k_ema<<<dim3((NB * Dm + 255) / 256), ET, 0, stream>>>(x1, ema);
    gemm(x1_bf, Dm, p2s_wh, Dm, p2s_b, pg, nullptr, Dm, MT, Dm, Dm, 0);
    k_cs<<<EW, ET, 0, stream>>>(x1, pg, ema, cs);
    gemm(cs, 2 * Dm, t1_wh, 2 * Dm, t1_b, nullptr, h_bf, 2 * Dm, MT, 2 * Dm, 2 * Dm, 1);
    gemm(h_bf, 2 * Dm, t2_wh, 2 * Dm, t2_b, deltas, nullptr, Dm, MT, Dm, 2 * Dm, 0);
    k_cumsum<<<dim3((NB * Dm + 255) / 256), ET, 0, stream>>>(seqstate, deltas, states);
    k_ln<<<dim3(MT), ET, 0, stream>>>(states, Dm, nullptr, ln3_g, ln3_b, states_ln, states_bf, Dm);
    gemm(states_bf, Dm, s2p_wh, Dm, s2p_b, sg, nullptr, Dm, MT, Dm, Dm, 0);

    // FFN + LN2 (final x output)
    k_fi<<<EW, ET, 0, stream>>>(x1, sg, fi_bf);
    gemm(fi_bf, Dm, ffn1_wh, Dm, ffn1_b, nullptr, hf_bf, 4 * Dm, MT, 4 * Dm, Dm, 1);
    gemm(hf_bf, 4 * Dm, ffn2_wh, 4 * Dm, ffn2_b, ffn, nullptr, Dm, MT, Dm, 4 * Dm, 0);
    k_ln<<<dim3(MT), ET, 0, stream>>>(ffn, Dm, x1, ln2_g, ln2_b, dout, nullptr, Dm);

    // final sequential state (LN4 of states[:, -1]) and trajectory summary
    k_ln<<<dim3(NB), ET, 0, stream>>>(states_ln + (size_t)(TT - 1) * Dm, (long)TT * Dm,
                                      nullptr, ln4_g, ln4_b, dout + NTD, nullptr, Dm);
    k_traj<<<dim3((NB * Dm + 255) / 256), ET, 0, stream>>>(ema, dout + NTD + (size_t)NB * Dm);

    (void)in_sizes; (void)n_in; (void)out_size; (void)ws_size;
}
